// RadixLayer_22368189678017
// MI455X (gfx1250) — matrix operand order repaired and checked by reading, not compiled, not run on hardware
//
#include <hip/hip_runtime.h>

typedef __attribute__((ext_vector_type(16))) __bf16 v16bf;
typedef __attribute__((ext_vector_type(8)))  float  v8f;

#define M_DIM 8192
#define K_DIM 4096
#define N_DIM 4096

// ---------------------------------------------------------------------------
// fp32 -> (bf16 hi, bf16 lo) split using the hardware cvt path (RNE).
// x ~= hi + lo ; D = Ah*Bh + Ah*Bl + Al*Bh gives near-fp32 GEMM accuracy
// from bf16 WMMAs at ~3x the f32-WMMA (16x16x4) rate.
// ---------------------------------------------------------------------------
__device__ __forceinline__ void split2(float x, unsigned short& h, unsigned short& l) {
    __bf16 hb = (__bf16)x;                                   // v_cvt (RNE)
    unsigned short hu = __builtin_bit_cast(unsigned short, hb);
    float fh = __builtin_bit_cast(float, (unsigned int)hu << 16);  // exact upcast
    __bf16 lb = (__bf16)(x - fh);
    h = hu;
    l = __builtin_bit_cast(unsigned short, lb);
}

// ===========================================================================
// Pre-pass 1: A[8192,4096] fp32 -> Ah, Al bf16 planes (row-major, same layout)
// ===========================================================================
__global__ __launch_bounds__(256)
void convertA(const float* __restrict__ A,
              unsigned short* __restrict__ Ah,
              unsigned short* __restrict__ Al)
{
    size_t base = ((size_t)blockIdx.x * 256 + threadIdx.x) * 8;
    float4 x0 = *(const float4*)(A + base);
    float4 x1 = *(const float4*)(A + base + 4);
    float x[8] = {x0.x, x0.y, x0.z, x0.w, x1.x, x1.y, x1.z, x1.w};
    unsigned int hp[4], lp[4];
#pragma unroll
    for (int i = 0; i < 4; ++i) {
        unsigned short h0, l0, h1, l1;
        split2(x[2*i],   h0, l0);
        split2(x[2*i+1], h1, l1);
        hp[i] = (unsigned int)h0 | ((unsigned int)h1 << 16);
        lp[i] = (unsigned int)l0 | ((unsigned int)l1 << 16);
    }
    *(uint4*)(Ah + base) = make_uint4(hp[0], hp[1], hp[2], hp[3]);
    *(uint4*)(Al + base) = make_uint4(lp[0], lp[1], lp[2], lp[3]);
}

// ===========================================================================
// Pre-pass 2: B = kernel .* mask, fp32[k][n] -> bf16 planes TRANSPOSED [n][k]
// (GEMM B staging becomes a contiguous row copy matching the WMMA B layout
// where (k,k+1) pairs pack into one dword). 64x64 tile transpose via LDS.
// ===========================================================================
#define TB 64
__global__ __launch_bounds__(256)
void convertB(const float* __restrict__ W, const float* __restrict__ Msk,
              unsigned short* __restrict__ Bh,   // [N_DIM][K_DIM]
              unsigned short* __restrict__ Bl)
{
    __shared__ unsigned short TH[TB][TB + 2];
    __shared__ unsigned short TL[TB][TB + 2];
    const int n0 = blockIdx.x * TB;
    const int k0 = blockIdx.y * TB;
    const int t  = threadIdx.x;
    const int r  = t >> 2;          // 0..63
    const int c0 = (t & 3) * 16;    // 0,16,32,48

    const float* wp = W   + (size_t)(k0 + r) * N_DIM + n0 + c0;
    const float* mp = Msk + (size_t)(k0 + r) * N_DIM + n0 + c0;
#pragma unroll
    for (int i = 0; i < 4; ++i) {
        float4 w4 = ((const float4*)wp)[i];
        float4 m4 = ((const float4*)mp)[i];
        float xs[4] = {w4.x * m4.x, w4.y * m4.y, w4.z * m4.z, w4.w * m4.w};
#pragma unroll
        for (int e = 0; e < 4; ++e) {
            unsigned short h, l;
            split2(xs[e], h, l);
            TH[r][c0 + 4*i + e] = h;
            TL[r][c0 + 4*i + e] = l;
        }
    }
    __syncthreads();

    unsigned int hp[8], lp[8];
#pragma unroll
    for (int i = 0; i < 8; ++i) {
        unsigned short h0 = TH[c0 + 2*i][r], h1 = TH[c0 + 2*i + 1][r];
        unsigned short l0 = TL[c0 + 2*i][r], l1 = TL[c0 + 2*i + 1][r];
        hp[i] = (unsigned int)h0 | ((unsigned int)h1 << 16);
        lp[i] = (unsigned int)l0 | ((unsigned int)l1 << 16);
    }
    unsigned short* oh = Bh + (size_t)(n0 + r) * K_DIM + k0 + c0;
    unsigned short* ol = Bl + (size_t)(n0 + r) * K_DIM + k0 + c0;
    ((uint4*)oh)[0] = make_uint4(hp[0], hp[1], hp[2], hp[3]);
    ((uint4*)oh)[1] = make_uint4(hp[4], hp[5], hp[6], hp[7]);
    ((uint4*)ol)[0] = make_uint4(lp[0], lp[1], lp[2], lp[3]);
    ((uint4*)ol)[1] = make_uint4(lp[4], lp[5], lp[6], lp[7]);
}

// ===========================================================================
// Main GEMM: async global->LDS staging (ASYNCcnt) + v_wmma_f32_16x16x32_bf16.
// Block tile 128x64x32, double-buffered LDS, 8 waves each owning 32x32.
// Row stride 20 uints (80B): every LDS op 16B-aligned b128, <=2-way banks.
// No VGPR staging registers -> no spills.
// ===========================================================================
#define BM 128
#define BN 64
#define BK 32
#define LDSW 20

struct U8 { uint4 a, b; };   // 32B fragment container -> bit_cast to v16bf

__global__ __launch_bounds__(256)
void radix_wmma_gemm(const unsigned short* __restrict__ Ah,
                     const unsigned short* __restrict__ Al,
                     const unsigned short* __restrict__ Bh,
                     const unsigned short* __restrict__ Bl,
                     float* __restrict__ C)
{
    __shared__ __align__(16) unsigned int AsH[2][BM][LDSW];
    __shared__ __align__(16) unsigned int AsL[2][BM][LDSW];
    __shared__ __align__(16) unsigned int BsH[2][BN][LDSW];
    __shared__ __align__(16) unsigned int BsL[2][BN][LDSW];   // 61,440 B total

    const int tid  = threadIdx.x;
    const int lane = tid & 31;
    const int wave = tid >> 5;
    const int wm   = (wave & 3) * 32;
    const int wn   = (wave >> 2) * 32;
    const int blockN = blockIdx.x * BN;
    const int blockM = blockIdx.y * BM;

    // A staging: thread -> row tid/2 (0..127), 16 bf16 at k-offset (tid&1)*16
    const int aRow = tid >> 1;
    const int aOff = (tid & 1) * 16;
    // B staging: thread -> n-row tid/4 (0..63), 8 bf16 at k-offset (tid&3)*8
    const int bRow = tid >> 2;
    const int bOff = (tid & 3) * 8;

    const unsigned short* aHg = Ah + (size_t)(blockM + aRow) * K_DIM + aOff;
    const unsigned short* aLg = Al + (size_t)(blockM + aRow) * K_DIM + aOff;
    const unsigned short* bHg = Bh + (size_t)(blockN + bRow) * K_DIM + bOff;
    const unsigned short* bLg = Bl + (size_t)(blockN + bRow) * K_DIM + bOff;

    // Async global->LDS copy of one K-step tile into buffer s.
    // INST_OFFSET is added to BOTH the global and the LDS address (ISA 15.18),
    // so the second 16B chunk of each 32B A-row reuses the same address pair.
    auto stageAsync = [&](int s, int kb) {
        unsigned ldsAH = (unsigned)(size_t)&AsH[s][aRow][aOff / 2];
        unsigned ldsAL = (unsigned)(size_t)&AsL[s][aRow][aOff / 2];
        unsigned ldsBH = (unsigned)(size_t)&BsH[s][bRow][bOff / 2];
        unsigned ldsBL = (unsigned)(size_t)&BsL[s][bRow][bOff / 2];
        const unsigned short* gAH = aHg + kb * BK;
        const unsigned short* gAL = aLg + kb * BK;
        const unsigned short* gBH = bHg + kb * BK;
        const unsigned short* gBL = bLg + kb * BK;
        asm volatile(
            "global_load_async_to_lds_b128 %0, %4, off\n\t"
            "global_load_async_to_lds_b128 %0, %4, off offset:16\n\t"
            "global_load_async_to_lds_b128 %1, %5, off\n\t"
            "global_load_async_to_lds_b128 %1, %5, off offset:16\n\t"
            "global_load_async_to_lds_b128 %2, %6, off\n\t"
            "global_load_async_to_lds_b128 %3, %7, off"
            :
            : "v"(ldsAH), "v"(ldsAL), "v"(ldsBH), "v"(ldsBL),
              "v"(gAH), "v"(gAL), "v"(gBH), "v"(gBL)
            : "memory");
    };

    v8f acc[2][2];
    v8f zf = {0.f, 0.f, 0.f, 0.f, 0.f, 0.f, 0.f, 0.f};
    acc[0][0] = zf; acc[0][1] = zf; acc[1][0] = zf; acc[1][1] = zf;

    auto compute = [&](int s) {
        U8 fAH[2], fAL[2], fBH[2], fBL[2];
        const int row16 = lane & 15;
        // A 16-bit layout: lanes 0-15 K=0-7 (V0-3) / 16-23 (V4-7); lanes 16-31 +8
        const int akp = (lane < 16) ? 0 : 4;
        // B 16-bit layout: lanes 0-15 K=0-7 (V0-3) / 16-23 (V4-7); lanes 16-31 +8
        const int bkp = (lane < 16) ? 0 : 4;
#pragma unroll
        for (int i = 0; i < 2; ++i) {
            int m = wm + i * 16 + row16;
            fAH[i].a = *(const uint4*)&AsH[s][m][akp];
            fAH[i].b = *(const uint4*)&AsH[s][m][akp + 8];
            fAL[i].a = *(const uint4*)&AsL[s][m][akp];
            fAL[i].b = *(const uint4*)&AsL[s][m][akp + 8];
        }
#pragma unroll
        for (int j = 0; j < 2; ++j) {
            int n = wn + j * 16 + row16;
            fBH[j].a = *(const uint4*)&BsH[s][n][bkp];
            fBH[j].b = *(const uint4*)&BsH[s][n][bkp + 8];
            fBL[j].a = *(const uint4*)&BsL[s][n][bkp];
            fBL[j].b = *(const uint4*)&BsL[s][n][bkp + 8];
        }
#pragma unroll
        for (int i = 0; i < 2; ++i) {
            v16bf vAh = __builtin_bit_cast(v16bf, fAH[i]);
            v16bf vAl = __builtin_bit_cast(v16bf, fAL[i]);
#pragma unroll
            for (int j = 0; j < 2; ++j) {
                v16bf vBh = __builtin_bit_cast(v16bf, fBH[j]);
                v16bf vBl = __builtin_bit_cast(v16bf, fBL[j]);
                acc[i][j] = __builtin_amdgcn_wmma_f32_16x16x32_bf16(
                    false, vAh, false, vBh, (short)0, acc[i][j], false, false);
                acc[i][j] = __builtin_amdgcn_wmma_f32_16x16x32_bf16(
                    false, vAh, false, vBl, (short)0, acc[i][j], false, false);
                acc[i][j] = __builtin_amdgcn_wmma_f32_16x16x32_bf16(
                    false, vAl, false, vBh, (short)0, acc[i][j], false, false);
            }
        }
    };

    const int KSTEPS = K_DIM / BK;   // 128

    stageAsync(0, 0);
    asm volatile("s_wait_asynccnt 0x0" ::: "memory");
    __syncthreads();

    int s = 0;
    for (int kb = 0; kb < KSTEPS; ++kb) {
        if (kb + 1 < KSTEPS) stageAsync(s ^ 1, kb + 1);  // overlap copy with WMMAs
        compute(s);
        asm volatile("s_wait_asynccnt 0x0" ::: "memory"); // drain own async stores
        __syncthreads();                                  // publish buffer s^1
        s ^= 1;
    }

    // C/D layout: VGPR r -> M=r (lanes 0-15) / r+8 (lanes 16-31); N = lane&15
    float* cBase = C + (size_t)blockM * N_DIM + blockN;
    const int mHalf = (lane < 16) ? 0 : 8;
#pragma unroll
    for (int i = 0; i < 2; ++i) {
        int mRow0 = wm + i * 16 + mHalf;
#pragma unroll
        for (int j = 0; j < 2; ++j) {
            int n = wn + j * 16 + (lane & 15);
#pragma unroll
            for (int r = 0; r < 8; ++r) {
                cBase[(size_t)(mRow0 + r) * N_DIM + n] = acc[i][j][r];
            }
        }
    }
}

// ===========================================================================
// Fallback: fused conversion GEMM (used only if ws_size < 192 MB)
// ===========================================================================
#define FLDSW 18

__global__ __launch_bounds__(256)
void radix_wmma_gemm_fused(const float* __restrict__ A,
                           const float* __restrict__ W,
                           const float* __restrict__ Msk,
                           float* __restrict__ C)
{
    __shared__ unsigned int AsH[2][BM][FLDSW];
    __shared__ unsigned int AsL[2][BM][FLDSW];
    __shared__ unsigned int BsH[2][BN][FLDSW];
    __shared__ unsigned int BsL[2][BN][FLDSW];

    const int tid  = threadIdx.x;
    const int lane = tid & 31;
    const int wave = tid >> 5;
    const int wm   = (wave & 3) * 32;
    const int wn   = (wave >> 2) * 32;
    const int blockN = blockIdx.x * BN;
    const int blockM = blockIdx.y * BM;

    const int aRow = tid >> 1;
    const int aCol = (tid & 1) * 16;
    const int bRow = tid >> 3;
    const int bCol = (tid & 7) * 8;

    const float* aG = A   + (size_t)(blockM + aRow) * K_DIM + aCol;
    const float* wG = W   + (size_t)bRow * N_DIM + blockN + bCol;
    const float* mG = Msk + (size_t)bRow * N_DIM + blockN + bCol;

    float aR[16];
    float bR[8];

    auto loadG = [&](int kb) {
        const float4* ap = (const float4*)(aG + (size_t)kb * BK);
#pragma unroll
        for (int i = 0; i < 4; ++i) {
            float4 v = ap[i];
            aR[4*i+0] = v.x; aR[4*i+1] = v.y; aR[4*i+2] = v.z; aR[4*i+3] = v.w;
        }
        size_t bo = (size_t)kb * BK * N_DIM;
        const float4* wp = (const float4*)(wG + bo);
        const float4* mp = (const float4*)(mG + bo);
#pragma unroll
        for (int i = 0; i < 2; ++i) {
            float4 w4 = wp[i], m4 = mp[i];
            bR[4*i+0] = w4.x * m4.x; bR[4*i+1] = w4.y * m4.y;
            bR[4*i+2] = w4.z * m4.z; bR[4*i+3] = w4.w * m4.w;
        }
    };

    auto stageLDS = [&](int s) {
#pragma unroll
        for (int i = 0; i < 8; ++i) {
            unsigned short h0, l0, h1, l1;
            split2(aR[2*i],   h0, l0);
            split2(aR[2*i+1], h1, l1);
            AsH[s][aRow][aCol/2 + i] = (unsigned int)h0 | ((unsigned int)h1 << 16);
            AsL[s][aRow][aCol/2 + i] = (unsigned int)l0 | ((unsigned int)l1 << 16);
        }
        unsigned short* bh = (unsigned short*)&BsH[s][0][0];
        unsigned short* bl = (unsigned short*)&BsL[s][0][0];
#pragma unroll
        for (int i = 0; i < 8; ++i) {
            unsigned short h, l;
            split2(bR[i], h, l);
            int n = bCol + i;
            bh[n * (2*FLDSW) + bRow] = h;
            bl[n * (2*FLDSW) + bRow] = l;
        }
    };

    v8f acc[2][2];
    v8f zf = {0.f, 0.f, 0.f, 0.f, 0.f, 0.f, 0.f, 0.f};
    acc[0][0] = zf; acc[0][1] = zf; acc[1][0] = zf; acc[1][1] = zf;

    auto compute = [&](int s) {
        unsigned int ah[2][8], al[2][8], bh[2][8], bl[2][8];
        const int row16 = lane & 15;
        const int akp = (lane < 16) ? 0 : 4;
        const int bkp = (lane < 16) ? 0 : 4;
#pragma unroll
        for (int i = 0; i < 2; ++i) {
            int m = wm + i * 16 + row16;
#pragma unroll
            for (int j = 0; j < 4; ++j) {
                ah[i][j]   = AsH[s][m][akp + j];
                ah[i][4+j] = AsH[s][m][akp + 8 + j];
                al[i][j]   = AsL[s][m][akp + j];
                al[i][4+j] = AsL[s][m][akp + 8 + j];
            }
        }
#pragma unroll
        for (int i = 0; i < 2; ++i) {
            int n = wn + i * 16 + row16;
#pragma unroll
            for (int j = 0; j < 8; ++j) {
                bh[i][j] = BsH[s][n][bkp + j + (j & 4)];
                bl[i][j] = BsL[s][n][bkp + j + (j & 4)];
            }
        }
#pragma unroll
        for (int i = 0; i < 2; ++i) {
            v16bf vAh = __builtin_bit_cast(v16bf, ah[i]);
            v16bf vAl = __builtin_bit_cast(v16bf, al[i]);
#pragma unroll
            for (int j = 0; j < 2; ++j) {
                v16bf vBh = __builtin_bit_cast(v16bf, bh[j]);
                v16bf vBl = __builtin_bit_cast(v16bf, bl[j]);
                acc[i][j] = __builtin_amdgcn_wmma_f32_16x16x32_bf16(
                    false, vAh, false, vBh, (short)0, acc[i][j], false, false);
                acc[i][j] = __builtin_amdgcn_wmma_f32_16x16x32_bf16(
                    false, vAh, false, vBl, (short)0, acc[i][j], false, false);
                acc[i][j] = __builtin_amdgcn_wmma_f32_16x16x32_bf16(
                    false, vAl, false, vBh, (short)0, acc[i][j], false, false);
            }
        }
    };

    const int KSTEPS = K_DIM / BK;

    loadG(0);
    stageLDS(0);
    __syncthreads();

    int s = 0;
    for (int kb = 0; kb < KSTEPS; ++kb) {
        if (kb + 1 < KSTEPS) loadG(kb + 1);
        compute(s);
        if (kb + 1 < KSTEPS) stageLDS(s ^ 1);
        __syncthreads();
        s ^= 1;
    }

    float* cBase = C + (size_t)blockM * N_DIM + blockN;
    const int mHalf = (lane < 16) ? 0 : 8;
#pragma unroll
    for (int i = 0; i < 2; ++i) {
        int mRow0 = wm + i * 16 + mHalf;
#pragma unroll
        for (int j = 0; j < 2; ++j) {
            int n = wn + j * 16 + (lane & 15);
#pragma unroll
            for (int r = 0; r < 8; ++r) {
                cBase[(size_t)(mRow0 + r) * N_DIM + n] = acc[i][j][r];
            }
        }
    }
}

// ===========================================================================
extern "C" void kernel_launch(void* const* d_in, const int* in_sizes, int n_in,
                              void* d_out, int out_size, void* d_ws, size_t ws_size,
                              hipStream_t stream) {
    const float* A   = (const float*)d_in[0];   // [8192, 4096]
    const float* W   = (const float*)d_in[1];   // [4096, 4096]
    const float* Msk = (const float*)d_in[2];   // [4096, 4096]
    float* C = (float*)d_out;                   // [8192, 4096]

    const size_t elemsA = (size_t)M_DIM * K_DIM;
    const size_t elemsB = (size_t)K_DIM * N_DIM;
    const size_t need   = (elemsA * 2 + elemsB * 2) * sizeof(unsigned short); // 192 MB

    if (ws_size >= need) {
        unsigned short* Ah = (unsigned short*)d_ws;
        unsigned short* Al = Ah + elemsA;
        unsigned short* Bh = Al + elemsA;       // transposed [N][K]
        unsigned short* Bl = Bh + elemsB;

        convertA<<<dim3((unsigned)(elemsA / 8 / 256)), dim3(256), 0, stream>>>(A, Ah, Al);
        convertB<<<dim3(N_DIM / TB, K_DIM / TB), dim3(256), 0, stream>>>(W, Msk, Bh, Bl);
        radix_wmma_gemm<<<dim3(N_DIM / BN, M_DIM / BM), dim3(256), 0, stream>>>(Ah, Al, Bh, Bl, C);
    } else {
        radix_wmma_gemm_fused<<<dim3(N_DIM / BN, M_DIM / BM), dim3(256), 0, stream>>>(A, W, Msk, C);
    }
}
